// ManifoldAlignmentLoss_33938831573408
// MI455X (gfx1250) — compile-verified, matrix operand order checked by reading, not run on hardware
//
#include <hip/hip_runtime.h>
#include <hip/hip_bf16.h>

// ---------------------------------------------------------------------------
// ManifoldAlignmentLoss for MI455X (gfx1250, wave32, WMMA bf16 path)
//   B=256, D=512, N=50000, A=8, K_NEIGHBORS=5
// Pipeline:
//   1) prep_z:     L2-normalize z rows -> bf16, pack target attrs (2b fields)
//   2) prep_train: L2-normalize train rows -> bf16 (padded to 50048), pack attrs
//   3) gemm_topk:  128x64 tile per block; each wave computes a 2x2 grid of
//                  16x16 tiles (4 accumulators) -> 1KB loaded per WMMA,
//                  64x v_wmma_f32_16x16x32_bf16 per block k-loop
//   4) finalize:   merge per-block top5 partials, apply valid mask, reduce
// ---------------------------------------------------------------------------

typedef __attribute__((ext_vector_type(16))) __bf16 v16bf;
typedef __attribute__((ext_vector_type(8)))  float  v8f;

#define D_DIM 512
#define B_DIM 256
#define N_DIM 50000
#define NPAD  50048          // padded to multiple of 64
#define NBLK  782            // NPAD / 64
#define KNEI  5
#define NEG_INF (-1e30f)

struct BF16Frag { uint4 lo, hi; };   // 32 bytes == v16bf

__device__ __forceinline__ unsigned pack_attrs(const int* a) {
    unsigned p = 0u;
#pragma unroll
    for (int i = 0; i < 8; ++i) p |= ((unsigned)(a[i] & 3)) << (2 * i);
    return p;
}

__device__ __forceinline__ void top5_insert(float v, float (&t)[KNEI]) {
#pragma unroll
    for (int i = 0; i < KNEI; ++i) {
        if (v > t[i]) { float tmp = t[i]; t[i] = v; v = tmp; }
    }
}

__device__ __forceinline__ v16bf load_frag(const __bf16* p) {
    BF16Frag f;
    f.lo = *(const uint4*)(p);        // K = kbase   .. +8 (16 bytes)
    f.hi = *(const uint4*)(p + 16);   // K = kbase+16.. +8
    return __builtin_bit_cast(v16bf, f);
}

// --------------------------- prep: z (256 rows) -----------------------------
__global__ void __launch_bounds__(128)
prep_z_kernel(const float* __restrict__ z, const int* __restrict__ tattr,
              __bf16* __restrict__ zb, unsigned* __restrict__ zpack) {
    const int row = blockIdx.x;          // 0..255
    const int tid = threadIdx.x;         // 0..127, 4 floats each
    __shared__ float red[128];

    const float4 x = ((const float4*)(z + (size_t)row * D_DIM))[tid];
    red[tid] = x.x * x.x + x.y * x.y + x.z * x.z + x.w * x.w;
    __syncthreads();
#pragma unroll
    for (int s = 64; s > 0; s >>= 1) {
        if (tid < s) red[tid] += red[tid + s];
        __syncthreads();
    }
    const float inv = 1.0f / fmaxf(sqrtf(red[0]), 1e-12f);

    __bf16* o = zb + (size_t)row * D_DIM + tid * 4;
    o[0] = (__bf16)(x.x * inv);
    o[1] = (__bf16)(x.y * inv);
    o[2] = (__bf16)(x.z * inv);
    o[3] = (__bf16)(x.w * inv);

    if (tid == 0) zpack[row] = pack_attrs(tattr + row * 8);
}

// ------------------------ prep: train (50048 rows) --------------------------
__global__ void __launch_bounds__(128)
prep_train_kernel(const float* __restrict__ emb, const int* __restrict__ nattr,
                  __bf16* __restrict__ tb, unsigned* __restrict__ tpack) {
    const int row = blockIdx.x;          // 0..NPAD-1
    const int tid = threadIdx.x;
    const bool live = (row < N_DIM);
    __shared__ float red[128];

    float4 x = make_float4(0.f, 0.f, 0.f, 0.f);
    if (live) x = ((const float4*)(emb + (size_t)row * D_DIM))[tid];
    red[tid] = x.x * x.x + x.y * x.y + x.z * x.z + x.w * x.w;
    __syncthreads();
#pragma unroll
    for (int s = 64; s > 0; s >>= 1) {
        if (tid < s) red[tid] += red[tid + s];
        __syncthreads();
    }
    const float inv = live ? 1.0f / fmaxf(sqrtf(red[0]), 1e-12f) : 0.0f;

    __bf16* o = tb + (size_t)row * D_DIM + tid * 4;
    o[0] = (__bf16)(x.x * inv);
    o[1] = (__bf16)(x.y * inv);
    o[2] = (__bf16)(x.z * inv);
    o[3] = (__bf16)(x.w * inv);

    if (tid == 0) tpack[row] = live ? pack_attrs(nattr + row * 8) : 0xFFFFFFFFu;
}

// ----------------- main: WMMA GEMM + mask + per-block top5 ------------------
// grid = (NBLK, 2); block = 256 threads = 8 waves arranged 4 (b) x 2 (n).
// Block tile: 128 b-rows x 64 n-cols. Each wave owns a 32x32 subtile computed
// as a 2x2 grid of 16x16 WMMA accumulators: per K-step, 2 A-frags + 2 B-frags
// feed 4 WMMAs (1KB of loads per WMMA -> 16 FLOP/byte from L2).
__global__ void __launch_bounds__(256)
gemm_topk_kernel(const __bf16* __restrict__ zb, const __bf16* __restrict__ tb,
                 const unsigned* __restrict__ zpack, const unsigned* __restrict__ tpack,
                 float* __restrict__ partials, int* __restrict__ counts) {
    const int nb  = blockIdx.x;              // n tile index (64 wide)
    const int n0  = nb * 64;
    const int b0  = blockIdx.y * 128;        // b tile base
    const int tid = threadIdx.x;
    const int wave = tid >> 5;
    const int lane = tid & 31;
    const int wr = wave >> 1;                // 0..3  (32 b-rows each)
    const int wc = wave & 1;                 // 0..1  (32 n-cols each)
    const int l15 = lane & 15;

    // 16-bit A/B fragment layout (ISA 7.12.2): half-wave selects K chunk base.
    const int half = (lane < 16) ? 0 : 8;
    const size_t mRow = (size_t)(b0 + wr * 32 + l15);      // A: z row (M)
    const size_t nCol = (size_t)(n0 + wc * 32 + l15);      // B: train row (N)

    const __bf16* a0 = zb + mRow * D_DIM + half;
    const __bf16* a1 = a0 + (size_t)16 * D_DIM;            // +16 b-rows
    const __bf16* p0 = tb + nCol * D_DIM + half;
    const __bf16* p1 = p0 + (size_t)16 * D_DIM;            // +16 n-cols

    v8f acc00 = {}, acc01 = {}, acc10 = {}, acc11 = {};
#pragma unroll
    for (int k0 = 0; k0 < D_DIM; k0 += 32) {
        const v16bf A0 = load_frag(a0 + k0);
        const v16bf A1 = load_frag(a1 + k0);
        const v16bf B0 = load_frag(p0 + k0);
        const v16bf B1 = load_frag(p1 + k0);
        acc00 = __builtin_amdgcn_wmma_f32_16x16x32_bf16(false, A0, false, B0,
                                                        (short)0, acc00, false, false);
        acc01 = __builtin_amdgcn_wmma_f32_16x16x32_bf16(false, A0, false, B1,
                                                        (short)0, acc01, false, false);
        acc10 = __builtin_amdgcn_wmma_f32_16x16x32_bf16(false, A1, false, B0,
                                                        (short)0, acc10, false, false);
        acc11 = __builtin_amdgcn_wmma_f32_16x16x32_bf16(false, A1, false, B1,
                                                        (short)0, acc11, false, false);
    }

    // C/D layout: VGPR i -> M = i (+8 for upper half-wave), N = lane&15.
    __shared__ float sims[128 * 65];                       // +1 pad per row
    const int colB = wc * 32 + l15;
    const int rowB = wr * 32 + ((lane < 16) ? 0 : 8);
#pragma unroll
    for (int i = 0; i < 8; ++i) {
        sims[(rowB + i)      * 65 + colB]      = acc00[i];
        sims[(rowB + i)      * 65 + colB + 16] = acc01[i];
        sims[(rowB + 16 + i) * 65 + colB]      = acc10[i];
        sims[(rowB + 16 + i) * 65 + colB + 16] = acc11[i];
    }
    __syncthreads();

    // Scan phase: thread t<128 owns b-row t of the tile; apply attribute mask,
    // count matches, keep a sorted top-5 over the 64 masked candidates.
    if (tid < 128) {
        const int b = b0 + tid;
        const unsigned zp = zpack[b];
        float t5[KNEI] = {NEG_INF, NEG_INF, NEG_INF, NEG_INF, NEG_INF};
        int cnt = 0;
#pragma unroll 4
        for (int j = 0; j < 64; ++j) {
            const unsigned tp = tpack[n0 + j];
            const unsigned x  = zp ^ tp;
            const unsigned y  = (x | (x >> 1)) & 0x55555555u;
            const bool m = (__popc(y) <= 1);               // >= 7 of 8 match
            cnt += m ? 1 : 0;
            const float v = m ? sims[tid * 65 + j] : 0.0f; // masked_sim
            top5_insert(v, t5);
        }
        float* p = partials + ((size_t)b * NBLK + nb) * KNEI;
#pragma unroll
        for (int i = 0; i < KNEI; ++i) p[i] = t5[i];
        atomicAdd(&counts[b], cnt);
    }
}

// ----------------------------- finalize -------------------------------------
__global__ void __launch_bounds__(256)
finalize_kernel(const float* __restrict__ partials, const int* __restrict__ counts,
                float* __restrict__ out) {
    const int b = threadIdx.x;               // one thread per sample
    float t5[KNEI] = {NEG_INF, NEG_INF, NEG_INF, NEG_INF, NEG_INF};
    const float* p = partials + (size_t)b * NBLK * KNEI;
    for (int i = 0; i < NBLK * KNEI; ++i) top5_insert(p[i], t5);

    const float mean_top = (t5[0] + t5[1] + t5[2] + t5[3] + t5[4]) * (1.0f / KNEI);
    const float loss = 1.0f - mean_top;                    // mean(1 - topk)
    const float valid = (counts[b] >= KNEI) ? 1.0f : 0.0f;

    __shared__ float red[256];
    red[b] = loss * valid;
    __syncthreads();
#pragma unroll
    for (int s = 128; s > 0; s >>= 1) {
        if (b < s) red[b] += red[b + s];
        __syncthreads();
    }
    if (b == 0) out[0] = red[0] / (float)B_DIM;
}

// ----------------------------- launcher -------------------------------------
extern "C" void kernel_launch(void* const* d_in, const int* in_sizes, int n_in,
                              void* d_out, int out_size, void* d_ws, size_t ws_size,
                              hipStream_t stream) {
    const float* z     = (const float*)d_in[0];   // [256, 512]
    const int*   tattr = (const int*)d_in[1];     // [256, 8]
    const float* emb   = (const float*)d_in[2];   // [50000, 512]
    const int*   nattr = (const int*)d_in[3];     // [50000, 8]
    float* out = (float*)d_out;

    char* ws = (char*)d_ws;
    size_t off = 0;
    auto take = [&](size_t bytes) -> char* {
        char* p = ws + off;
        off = (off + bytes + 255) & ~(size_t)255;
        return p;
    };
    __bf16*   train_bf = (__bf16*)  take((size_t)NPAD * D_DIM * sizeof(__bf16)); // 51.2 MB
    __bf16*   z_bf     = (__bf16*)  take((size_t)B_DIM * D_DIM * sizeof(__bf16));
    unsigned* tpack    = (unsigned*)take((size_t)NPAD * sizeof(unsigned));
    unsigned* zpack    = (unsigned*)take((size_t)B_DIM * sizeof(unsigned));
    int*      counts   = (int*)     take((size_t)B_DIM * sizeof(int));
    float*    partials = (float*)   take((size_t)B_DIM * NBLK * KNEI * sizeof(float)); // 4 MB

    hipMemsetAsync(counts, 0, (size_t)B_DIM * sizeof(int), stream);

    prep_z_kernel<<<B_DIM, 128, 0, stream>>>(z, tattr, z_bf, zpack);
    prep_train_kernel<<<NPAD, 128, 0, stream>>>(emb, nattr, train_bf, tpack);
    gemm_topk_kernel<<<dim3(NBLK, 2), 256, 0, stream>>>(z_bf, train_bf, zpack, tpack,
                                                        partials, counts);
    finalize_kernel<<<1, 256, 0, stream>>>(partials, counts, out);
}
